// POTLoss_6485400617244
// MI455X (gfx1250) — matrix operand order repaired and checked by reading, not compiled, not run on hardware
//
#include <hip/hip_runtime.h>
#include <hip/hip_bf16.h>
#include <math.h>

#define Bn 1024
#define Dn 1024
#define EPS_REG 0.05f
#define M_MASS 0.95f
#define N_ITER 50

typedef __attribute__((ext_vector_type(16))) __bf16 v16bf;
typedef __attribute__((ext_vector_type(8)))  float  v8f;

// Split an f32 into exact bf16 "hi" (truncated mantissa prefix) + bf16 "lo" residual.
__device__ __forceinline__ void pack_hi_lo(const float* s, v16bf& h, v16bf& l) {
#pragma unroll
  for (int e = 0; e < 16; ++e) {
    float x  = s[e];
    float hf = __uint_as_float(__float_as_uint(x) & 0xFFFF0000u);
    h[e] = (__bf16)hf;          // exact (low mantissa already zero)
    l[e] = (__bf16)(x - hf);    // residual
  }
}

__device__ __forceinline__ v8f wmma_bf16(v16bf a, v16bf b, v8f c) {
  // (neg_a, A, neg_b, B, c_mod, C, reuse_a, reuse_b)
  return __builtin_amdgcn_wmma_f32_16x16x32_bf16(false, a, false, b, (short)0, c,
                                                 false, false);
}

// ---------------------------------------------------------------- norms
__global__ void norms_kernel(const float* __restrict__ A, const float* __restrict__ T,
                             float* __restrict__ na, float* __restrict__ nb) {
  __shared__ float red[256];
  const int i = blockIdx.x;
  const float* src = blockIdx.y ? T : A;
  float* dst       = blockIdx.y ? nb : na;
  float s = 0.f;
  for (int d = threadIdx.x; d < Dn; d += 256) {
    float x = src[(size_t)i * Dn + d];
    s += x * x;
  }
  red[threadIdx.x] = s; __syncthreads();
  for (int st = 128; st > 0; st >>= 1) {
    if (threadIdx.x < st) red[threadIdx.x] += red[threadIdx.x + st];
    __syncthreads();
  }
  if (threadIdx.x == 0) dst[i] = red[0];
}

// ---------------------------------------------------------------- init
__global__ void init_kernel(float* u, float* v, float* p, float* q,
                            unsigned int* maxbits) {
  int i = blockIdx.x * 256 + threadIdx.x;
  if (i < Bn) { u[i] = 1.f; v[i] = 1.f; p[i] = 1.f; q[i] = 1.f; }
  if (i == 0) *maxbits = 0u;
}

// ---------------------------------------------------------------- cost GEMM (WMMA bf16x3)
// Grid (16,16), 256 threads (8 waves). Block tile 64x64; waves 4x2, each wave
// owns a 16x32 strip (two 16x16 WMMA accumulators sharing the A fragment).
__global__ void __launch_bounds__(256)
gemm_cost_kernel(const float* __restrict__ A, const float* __restrict__ T,
                 const float* __restrict__ na, const float* __restrict__ nb,
                 float* __restrict__ Mout, unsigned int* __restrict__ maxbits) {
  __shared__ float red[256];
  const int lane = threadIdx.x & 31;
  const int wave = threadIdx.x >> 5;
  const int half = lane >> 4;     // which 16-lane half
  const int l15  = lane & 15;
  const int wr   = wave >> 1;     // 0..3 (row of wave grid)
  const int wc   = wave & 1;      // 0..1 (col of wave grid)
  const int i0   = blockIdx.y * 64 + wr * 16;
  const int j0   = blockIdx.x * 64 + wc * 32;

  const float* arow  = A + (size_t)(i0 + l15) * Dn;
  const float* trow0 = T + (size_t)(j0 + l15) * Dn;
  const float* trow1 = trow0 + (size_t)16 * Dn;

  v8f acc0 = {}; v8f acc1 = {};
  for (int k0 = 0; k0 < Dn; k0 += 32) {
    // A fragment 16x32 bf16 layout: elems 0..15 = K=half*16+0..15 (contiguous)
    float abuf[16];
    const float* ap0 = arow + k0 + half * 16;
    const float* ap1 = arow + k0 + half * 16 + 8;
#pragma unroll
    for (int e = 0; e < 8; ++e) abuf[e]     = ap0[e];
#pragma unroll
    for (int e = 0; e < 8; ++e) abuf[8 + e] = ap1[e];
    v16bf ah, al; pack_hi_lo(abuf, ah, al);

    // B fragment 32x16 bf16 layout: lane = column, elems = K=half*16+0..15 (contiguous)
    float bbuf[16];
    const float* bp0 = trow0 + k0 + half * 16;
#pragma unroll
    for (int e = 0; e < 16; ++e) bbuf[e] = bp0[e];
    v16bf bh, bl; pack_hi_lo(bbuf, bh, bl);
    acc0 = wmma_bf16(ah, bh, acc0);
    acc0 = wmma_bf16(al, bh, acc0);
    acc0 = wmma_bf16(ah, bl, acc0);

    const float* bp1 = trow1 + k0 + half * 16;
#pragma unroll
    for (int e = 0; e < 16; ++e) bbuf[e] = bp1[e];
    pack_hi_lo(bbuf, bh, bl);
    acc1 = wmma_bf16(ah, bh, acc1);
    acc1 = wmma_bf16(al, bh, acc1);
    acc1 = wmma_bf16(ah, bl, acc1);
  }

  // C/D layout: VGPR v, lanes<16 -> M=v, lanes>=16 -> M=v+8; N = lane&15.
  float lmax = 0.f;
#pragma unroll
  for (int nt = 0; nt < 2; ++nt) {
#pragma unroll
    for (int vv = 0; vv < 8; ++vv) {
      int gi = i0 + vv + 8 * half;
      int gj = j0 + nt * 16 + l15;
      float dotv = nt ? acc1[vv] : acc0[vv];
      float m = fmaxf(na[gi] + nb[gj] - 2.f * dotv, 0.f);
      Mout[(size_t)gi * Bn + gj] = m;
      lmax = fmaxf(lmax, m);
    }
  }
  red[threadIdx.x] = lmax; __syncthreads();
  for (int st = 128; st > 0; st >>= 1) {
    if (threadIdx.x < st)
      red[threadIdx.x] = fmaxf(red[threadIdx.x], red[threadIdx.x + st]);
    __syncthreads();
  }
  if (threadIdx.x == 0) atomicMax(maxbits, __float_as_uint(red[0]));  // exact, order-free
}

// ---------------------------------------------------------------- Kt = exp(-M/(reg*maxM)), rowsums
__global__ void kexp_kernel(float* __restrict__ Kt, const unsigned int* maxbits,
                            float* __restrict__ rowred) {
  __shared__ float red[256];
  const int i = blockIdx.x;
  const float inv = 1.f / (EPS_REG * __uint_as_float(*maxbits));
  float s = 0.f;
  for (int j = threadIdx.x; j < Bn; j += 256) {
    float k = expf(-Kt[(size_t)i * Bn + j] * inv);
    Kt[(size_t)i * Bn + j] = k;
    s += k;
  }
  red[threadIdx.x] = s; __syncthreads();
  for (int st = 128; st > 0; st >>= 1) {
    if (threadIdx.x < st) red[threadIdx.x] += red[threadIdx.x + st];
    __syncthreads();
  }
  if (threadIdx.x == 0) rowred[i] = red[0];
}

__global__ void sumfin_kernel(const float* __restrict__ rowred, float* gs) {
  __shared__ float red[256];
  float s = 0.f;
  for (int i = threadIdx.x; i < Bn; i += 256) s += rowred[i];
  red[threadIdx.x] = s; __syncthreads();
  for (int st = 128; st > 0; st >>= 1) {
    if (threadIdx.x < st) red[threadIdx.x] += red[threadIdx.x + st];
    __syncthreads();
  }
  if (threadIdx.x == 0) *gs = M_MASS / red[0];  // K0 = Kt * m/sum(Kt)
}

// ---------------------------------------------------------------- Dykstra row step: y = Kt*v
__global__ void row_kernel(const float* __restrict__ Kt, const float* __restrict__ v,
                           float* __restrict__ u, float* __restrict__ p,
                           const float* __restrict__ gs) {
  __shared__ float red[256];
  const int i = blockIdx.x;
  float s = 0.f;
  for (int j = threadIdx.x; j < Bn; j += 256) s += Kt[(size_t)i * Bn + j] * v[j];
  red[threadIdx.x] = s; __syncthreads();
  for (int st = 128; st > 0; st >>= 1) {
    if (threadIdx.x < st) red[threadIdx.x] += red[threadIdx.x + st];
    __syncthreads();
  }
  if (threadIdx.x == 0) {
    float up = u[i] * p[i];                       // row factor of Kp (q1 applied)
    float rowsum = (*gs) * up * red[0];
    float r = fminf((1.0f / Bn) / rowsum, 1.f);   // a_i = 1/B
    u[i] = up * r;                                // K1 row factor
    p[i] = 1.f / r;                               // q1_new
  }
}

// ---------------------------------------------------------------- col step partials: zp = partial Kt^T * u
__global__ void colpart_kernel(const float* __restrict__ Kt, const float* __restrict__ u,
                               float* __restrict__ zp) {
  const int col = blockIdx.x * 256 + threadIdx.x;
  const int r0  = blockIdx.y * 128;
  float s = 0.f;
  for (int r = r0; r < r0 + 128; ++r) s += Kt[(size_t)r * Bn + col] * u[r];
  zp[(size_t)blockIdx.y * Bn + col] = s;
}

// ---------------------------------------------------------------- col finalize + dot(z, v_new)
__global__ void colfin_kernel(const float* __restrict__ zp, float* __restrict__ v,
                              float* __restrict__ q, const float* __restrict__ gs,
                              float* __restrict__ dotp) {
  __shared__ float red[256];
  const int col = blockIdx.x * 256 + threadIdx.x;
  float z = 0.f;
#pragma unroll
  for (int c = 0; c < 8; ++c) z += zp[(size_t)c * Bn + col];  // deterministic combine
  float vq = v[col] * q[col];                                 // col factor of K1p
  float colsum = vq * (*gs) * z;
  float c = fminf((1.0f / Bn) / colsum, 1.f);                 // b_j = 1/B
  float vn = vq * c;
  v[col] = vn;                                                // K2 col factor
  q[col] = 1.f / c;                                           // q2_new
  red[threadIdx.x] = z * vn; __syncthreads();
  for (int st = 128; st > 0; st >>= 1) {
    if (threadIdx.x < st) red[threadIdx.x] += red[threadIdx.x + st];
    __syncthreads();
  }
  if (threadIdx.x == 0) dotp[blockIdx.x] = red[0];
}

// ---------------------------------------------------------------- mass renormalize: gs = m/dot
__global__ void mass_kernel(const float* __restrict__ dotp, float* gs) {
  if (threadIdx.x == 0) {
    float d = ((dotp[0] + dotp[1]) + (dotp[2] + dotp[3]));
    *gs = M_MASS / d;   // sum(K2)=gs_old*dot; Kn scale = m/sum -> new global scale m/dot
  }
}

// ---------------------------------------------------------------- per-row log-softmax diagonal
__global__ void loss_row_kernel(const float* __restrict__ Kt, const float* __restrict__ u,
                                const float* __restrict__ v, const float* __restrict__ gs,
                                float* __restrict__ rowred) {
  __shared__ float sp[Bn];
  __shared__ float red[256];
  const int i = blockIdx.x;
  const float ru = (*gs) * u[i];
  float lm = -1e30f;
  for (int j = threadIdx.x; j < Bn; j += 256) {
    float pv = Kt[(size_t)i * Bn + j] * ru * v[j];
    sp[j] = pv;
    lm = fmaxf(lm, pv);
  }
  red[threadIdx.x] = lm; __syncthreads();
  for (int st = 128; st > 0; st >>= 1) {
    if (threadIdx.x < st)
      red[threadIdx.x] = fmaxf(red[threadIdx.x], red[threadIdx.x + st]);
    __syncthreads();
  }
  const float rmax = red[0]; __syncthreads();
  float se = 0.f;
  for (int j = threadIdx.x; j < Bn; j += 256) se += expf(sp[j] - rmax);
  red[threadIdx.x] = se; __syncthreads();
  for (int st = 128; st > 0; st >>= 1) {
    if (threadIdx.x < st) red[threadIdx.x] += red[threadIdx.x + st];
    __syncthreads();
  }
  if (threadIdx.x == 0) rowred[i] = sp[i] - rmax - logf(red[0]);  // logp[i,i]
}

__global__ void loss_final_kernel(const float* __restrict__ rowred, float* __restrict__ out) {
  __shared__ float red[256];
  float s = 0.f;
  for (int i = threadIdx.x; i < Bn; i += 256) s += rowred[i];
  red[threadIdx.x] = s; __syncthreads();
  for (int st = 128; st > 0; st >>= 1) {
    if (threadIdx.x < st) red[threadIdx.x] += red[threadIdx.x + st];
    __syncthreads();
  }
  if (threadIdx.x == 0) out[0] = -red[0] / (float)Bn;
}

// ----------------------------------------------------------------
extern "C" void kernel_launch(void* const* d_in, const int* in_sizes, int n_in,
                              void* d_out, int out_size, void* d_ws, size_t ws_size,
                              hipStream_t stream) {
  (void)in_sizes; (void)n_in; (void)out_size; (void)ws_size;
  const float* A = (const float*)d_in[0];  // audio_emb [1024,1024] f32
  const float* T = (const float*)d_in[1];  // text_emb  [1024,1024] f32
  // d_in[2] (labels) is unused by the loss math.

  float* ws     = (float*)d_ws;
  float* Kt     = ws;                    // 1M floats: Mraw then Kt=exp(-M') in place
  float* na     = ws + (1u << 20);
  float* nb     = na + Bn;
  float* u      = nb + Bn;
  float* v      = u + Bn;
  float* p      = v + Bn;
  float* q      = p + Bn;
  float* zp     = q + Bn;                // 8*Bn column partials
  float* rowred = zp + 8 * Bn;           // Bn
  float* dotp   = rowred + Bn;           // 4
  float* gs     = dotp + 4;              // global scale scalar
  unsigned int* maxbits = (unsigned int*)(gs + 1);

  init_kernel<<<dim3(4), dim3(256), 0, stream>>>(u, v, p, q, maxbits);
  norms_kernel<<<dim3(Bn, 2), dim3(256), 0, stream>>>(A, T, na, nb);
  gemm_cost_kernel<<<dim3(16, 16), dim3(256), 0, stream>>>(A, T, na, nb, Kt, maxbits);
  kexp_kernel<<<dim3(Bn), dim3(256), 0, stream>>>(Kt, maxbits, rowred);
  sumfin_kernel<<<1, 256, 0, stream>>>(rowred, gs);

  for (int it = 0; it < N_ITER; ++it) {
    row_kernel<<<dim3(Bn), dim3(256), 0, stream>>>(Kt, v, u, p, gs);
    colpart_kernel<<<dim3(4, 8), dim3(256), 0, stream>>>(Kt, u, zp);
    colfin_kernel<<<dim3(4), dim3(256), 0, stream>>>(zp, v, q, gs, dotp);
    mass_kernel<<<1, 32, 0, stream>>>(dotp, gs);
  }

  loss_row_kernel<<<dim3(Bn), dim3(256), 0, stream>>>(Kt, u, v, gs, rowred);
  loss_final_kernel<<<1, 256, 0, stream>>>(rowred, (float*)d_out);
}
